// SelfAttention_52604759441854
// MI455X (gfx1250) — compile-verified, matrix operand order checked by reading, not run on hardware
//
#include <hip/hip_runtime.h>

typedef __attribute__((ext_vector_type(16))) _Float16 v16h;
typedef __attribute__((ext_vector_type(8)))  float    v8f;
typedef __attribute__((ext_vector_type(4)))  unsigned int v4u;
typedef __attribute__((ext_vector_type(4)))  int      v4i;
typedef __attribute__((ext_vector_type(8)))  int      v8i;

#define WMMA_F16(a, b, c) \
  __builtin_amdgcn_wmma_f32_16x16x32_f16(false, (a), false, (b), (short)0, (c), false, false)

#if __has_builtin(__builtin_amdgcn_tensor_load_to_lds) && __has_builtin(__builtin_amdgcn_s_wait_tensorcnt)
#define HAVE_TDM 1
#else
#define HAVE_TDM 0
#endif

// POD 16-byte chunk of 8 f16 (avoid HIP float4 ctor-in-union issues)
struct alignas(16) f16x8 { _Float16 h[8]; };

__device__ __forceinline__ v16h frag_ld(const _Float16* p0, const _Float16* p1) {
  union { v16h v; f16x8 g[2]; } u;
  u.g[0] = *reinterpret_cast<const f16x8*>(p0);
  u.g[1] = *reinterpret_cast<const f16x8*>(p1);
  return u.v;
}

__device__ __forceinline__ void store8h(_Float16* dst, v8f a) {
  f16x8 t;
#pragma unroll
  for (int r = 0; r < 8; ++r) t.h[r] = (_Float16)a[r];
  *reinterpret_cast<f16x8*>(dst) = t;
}

#if HAVE_TDM
// Issue one TDM 2D tile load: global (row-major, dim0 contiguous, f16) -> LDS
// (packed rows). Descriptor per CDNA5 ISA ch.8: group0 = {count, lds_addr,
// global_addr[56:0], type=2}; group1 = {data_size=2B, tensor dims, tile dims,
// dim0 stride}. groups 2/3 zero (2D tensor). Wave-level op, EXEC ignored.
__device__ __forceinline__ void tdm_load_2d(unsigned lds_off, const void* gptr,
                                            unsigned tile_d0, unsigned tile_d1,
                                            unsigned long long stride_elems) {
  unsigned long long ga = (unsigned long long)gptr;
  const unsigned td0 = 1u << 20, td1 = 1u << 20;   // generous tensor bounds (no OOB clip)
  v4u g0;
  g0[0] = 1u;                                       // count=1, user descriptor
  g0[1] = lds_off;                                  // LDS byte address
  g0[2] = (unsigned)(ga & 0xffffffffu);             // global_addr[31:0]
  g0[3] = (unsigned)((ga >> 32) & 0x1ffffffu) | 0x80000000u;  // [56:32] | type=2
  v8i g1;
  g1[0] = (int)(1u << 16);                          // workgroup_mask=0, data_size=1 (2 bytes)
  g1[1] = (int)((td0 & 0xffffu) << 16);             // [63:48] = tensor_dim0[15:0]
  g1[2] = (int)((td0 >> 16) | ((td1 & 0xffffu) << 16));
  g1[3] = (int)((td1 >> 16) | (tile_d0 << 16));     // [127:112] = tile_dim0
  g1[4] = (int)(tile_d1 & 0xffffu);                 // tile_dim1 (tile_dim2=0)
  g1[5] = (int)(unsigned)(stride_elems & 0xffffffffu);       // dim0 stride lo32
  g1[6] = (int)(unsigned)((stride_elems >> 32) & 0xffffu);   // dim0 stride hi16
  g1[7] = 0;
  v4i gz = {0, 0, 0, 0};
#if defined(__clang_major__) && (__clang_major__ >= 23)
  v8i gz8 = {0, 0, 0, 0, 0, 0, 0, 0};
  __builtin_amdgcn_tensor_load_to_lds(g0, g1, gz, gz, gz8, 0);
#else
  __builtin_amdgcn_tensor_load_to_lds(g0, g1, gz, gz, 0);
#endif
}
#endif

// ---------------------------------------------------------------------------
// Kernel 1: convert/pack weights to f16.
// Wc[384][512] = rows 0..63 theta_w, 64..127 phi_w, 128..383 g_w.
// oWh[512][256] = o_w.
// ---------------------------------------------------------------------------
__global__ void k_convert(const float* __restrict__ theta_w, const float* __restrict__ phi_w,
                          const float* __restrict__ g_w, const float* __restrict__ o_w,
                          _Float16* __restrict__ Wc, _Float16* __restrict__ oWh) {
  int i = blockIdx.x * 256 + threadIdx.x;
  if (i < 384 * 512) {
    int f = i >> 9, c = i & 511;
    float v = (f < 64) ? theta_w[f * 512 + c]
            : (f < 128) ? phi_w[(f - 64) * 512 + c]
                        : g_w[(f - 128) * 512 + c];
    Wc[i] = (_Float16)v;
  }
  if (i < 512 * 256) oWh[i] = (_Float16)o_w[i];
}

// ---------------------------------------------------------------------------
// Kernel 2: fused projection GEMM. Per (batch, n-tile of 128 pixels):
//   Y[384 x 128] = Wc[384 x 512] * x_b[512 x 128]
// Rows 0..63 -> theta stored transposed [B][4096][64] (f16).
// Rows 64..383 -> pgraw [B][320][4096] (f16, pre-pool).
// 512 threads = 16 waves: 8 M-strips (48 rows) x 2 N-halves (64 cols).
// ---------------------------------------------------------------------------
__global__ __launch_bounds__(512) void k_proj(const float* __restrict__ x,
                                              const _Float16* __restrict__ Wc,
                                              _Float16* __restrict__ theta,
                                              _Float16* __restrict__ pgraw) {
  __shared__ alignas(16) _Float16 xT[128 * 40];   // [n][k] transposed, 16B-aligned rows
  const int b = blockIdx.y;
  const int n0 = blockIdx.x * 128;
  const int tid = threadIdx.x;
  const int lane = tid & 31, wave = tid >> 5;
  const int m0 = (wave >> 1) * 48;
  const int nb = (wave & 1) * 64;
  const int lrow = lane & 15;
  const int kb0 = (lane < 16) ? 0 : 8;

  const v8f vzero = {0, 0, 0, 0, 0, 0, 0, 0};
  v8f acc[3][4];
#pragma unroll
  for (int mt = 0; mt < 3; ++mt)
#pragma unroll
    for (int nt = 0; nt < 4; ++nt) acc[mt][nt] = vzero;

  for (int kc = 0; kc < 16; ++kc) {
    const int c0 = kc * 32;
    __syncthreads();
#pragma unroll
    for (int i = 0; i < 8; ++i) {      // stage 32k x 128n f32 -> f16 transposed
      int e = tid + i * 512;
      int kk = e >> 7, nn = e & 127;
      float v = x[((size_t)b * 512 + c0 + kk) * 4096 + n0 + nn];
      xT[nn * 40 + kk] = (_Float16)v;
    }
    if (kc + 1 < 16)                   // global_prefetch_b8 of next k-chunk
      __builtin_prefetch(&x[((size_t)b * 512 + c0 + 32 + (tid >> 4)) * 4096 + n0 + (tid & 15) * 8], 0, 0);
    __syncthreads();

    v16h bf[4];
#pragma unroll
    for (int nt = 0; nt < 4; ++nt) {
      const _Float16* p = &xT[(nb + nt * 16 + lrow) * 40];
      bf[nt] = frag_ld(p + kb0, p + kb0 + 16);
    }
#pragma unroll
    for (int mt = 0; mt < 3; ++mt) {
      const _Float16* p = &Wc[(m0 + mt * 16 + lrow) * 512 + c0];
      v16h af = frag_ld(p + kb0, p + kb0 + 16);
#pragma unroll
      for (int nt = 0; nt < 4; ++nt) acc[mt][nt] = WMMA_F16(af, bf[nt], acc[mt][nt]);
    }
  }

#pragma unroll
  for (int mt = 0; mt < 3; ++mt) {
    const int fbase = m0 + mt * 16 + kb0;   // 8 consecutive feature rows per lane
#pragma unroll
    for (int nt = 0; nt < 4; ++nt) {
      const int n = n0 + nb + nt * 16 + lrow;
      if (fbase < 64) {
        store8h(&theta[((size_t)b * 4096 + n) * 64 + fbase], acc[mt][nt]);
      } else {
#pragma unroll
        for (int r = 0; r < 8; ++r)
          pgraw[((size_t)b * 320 + (fbase - 64) + r) * 4096 + n] = (_Float16)acc[mt][nt][r];
      }
    }
  }
}

// ---------------------------------------------------------------------------
// Kernel 3: 2x2 max pool. phiT[B][1024][64] (transposed for S B-frags),
// gP[B][256][1024] (natural layout == O B-frags).
// ---------------------------------------------------------------------------
__global__ void k_pool(const _Float16* __restrict__ pgraw,
                       _Float16* __restrict__ phiT, _Float16* __restrict__ gP) {
  int i = blockIdx.x * 256 + threadIdx.x;
  if (i >= 16 * 320 * 1024) return;
  int m = i & 1023;
  int f = (i >> 10) % 320;
  int b = i / (320 * 1024);
  int mr = m >> 5, mw = m & 31;
  size_t base = ((size_t)b * 320 + f) * 4096 + (size_t)(2 * mr) * 64 + 2 * mw;
  float v = fmaxf(fmaxf((float)pgraw[base], (float)pgraw[base + 1]),
                  fmaxf((float)pgraw[base + 64], (float)pgraw[base + 65]));
  if (f < 64) phiT[((size_t)b * 1024 + m) * 64 + f] = (_Float16)v;
  else        gP[((size_t)b * 256 + (f - 64)) * 1024 + m] = (_Float16)v;
}

// ---------------------------------------------------------------------------
// Kernel 4: flash attention. 64-query tile per block, KV = 1024 in blocks of 64.
// 256 threads = 8 waves: 4 q-strips x 2 output-channel halves (128 each).
// K/V blocks are DMA'd to LDS by the Tensor Data Mover (one tensor_load_to_lds
// per block-iteration, wave 0, s_wait_tensorcnt + barrier), then all 8 waves
// feed WMMA from LDS. S = Q x phiBlk; online softmax; O += P x V.
// ---------------------------------------------------------------------------
__global__ __launch_bounds__(256) void k_attn(const _Float16* __restrict__ theta,
                                              const _Float16* __restrict__ phiT,
                                              const _Float16* __restrict__ gP,
                                              _Float16* __restrict__ o_in) {
  __shared__ alignas(16) _Float16 P[64 * 72];     // P tile [qrow][m]
#if HAVE_TDM
  __shared__ alignas(16) _Float16 phiL[64 * 64];  // K block  [m_local][d]   (8 KB)
  __shared__ alignas(16) _Float16 VL[256 * 64];   // V block  [c][m_local]   (32 KB)
#endif
  const int b = blockIdx.y;
  const int q0 = blockIdx.x * 64;
  const int tid = threadIdx.x;
  const int lane = tid & 31, wave = tid >> 5;
  const int strip = (wave >> 1) * 16;   // q-row strip
  const int cbase = (wave & 1) * 128;   // output-channel half
  const int lrow = lane & 15;
  const int kb0 = (lane < 16) ? 0 : 8;
  const int rowoff = (lane < 16) ? 0 : 8;

  v16h aq[2];
  {
    const _Float16* p = &theta[((size_t)b * 4096 + q0 + strip + lrow) * 64];
    aq[0] = frag_ld(p + kb0, p + kb0 + 16);
    aq[1] = frag_ld(p + 32 + kb0, p + 32 + kb0 + 16);
  }

  const v8f vzero = {0, 0, 0, 0, 0, 0, 0, 0};
  float rmax[8], rsum[8];
  v8f oacc[8];
#pragma unroll
  for (int r = 0; r < 8; ++r) { rmax[r] = -1e30f; rsum[r] = 0.f; }
#pragma unroll
  for (int ct = 0; ct < 8; ++ct) oacc[ct] = vzero;

  for (int j = 0; j < 16; ++j) {
    const int mb = j * 64;

#if HAVE_TDM
    __syncthreads();                  // prior-iteration LDS reads complete
    if (wave == 0) {                  // wave-uniform branch: one TDM issue per block
      // K block: 64 rows of phiT (contiguous 64x64 f16) -> phiL
      tdm_load_2d((unsigned)(size_t)&phiL[0],
                  &phiT[((size_t)b * 1024 + mb) * 64], 4096u, 1u, 4096ull);
      // V block: 256 rows x 64 f16, row stride 1024 -> VL packed rows of 64
      tdm_load_2d((unsigned)(size_t)&VL[0],
                  &gP[(size_t)b * 256 * 1024 + mb], 64u, 256u, 1024ull);
      __builtin_amdgcn_s_wait_tensorcnt(0);
    }
    __syncthreads();                  // TDM data visible to all waves
#endif

    v8f s[4] = {vzero, vzero, vzero, vzero};
#pragma unroll
    for (int kc = 0; kc < 2; ++kc)
#pragma unroll
      for (int mt = 0; mt < 4; ++mt) {
#if HAVE_TDM
        const _Float16* p = &phiL[(mt * 16 + lrow) * 64 + kc * 32];
#else
        const _Float16* p = &phiT[((size_t)b * 1024 + mb + mt * 16 + lrow) * 64 + kc * 32];
#endif
        v16h bf = frag_ld(p + kb0, p + kb0 + 16);
        s[mt] = WMMA_F16(aq[kc], bf, s[mt]);
      }

    float scale[8];
#pragma unroll
    for (int r = 0; r < 8; ++r) {
      float bm = fmaxf(fmaxf(s[0][r], s[1][r]), fmaxf(s[2][r], s[3][r]));
      bm = fmaxf(bm, __shfl_xor(bm, 1, 32));
      bm = fmaxf(bm, __shfl_xor(bm, 2, 32));
      bm = fmaxf(bm, __shfl_xor(bm, 4, 32));
      bm = fmaxf(bm, __shfl_xor(bm, 8, 32));
      float nm = fmaxf(rmax[r], bm);
      scale[r] = __expf(rmax[r] - nm);
      rmax[r] = nm;
      float ps = 0.f;
#pragma unroll
      for (int mt = 0; mt < 4; ++mt) {
        float e = __expf(s[mt][r] - nm);
        s[mt][r] = e;
        ps += e;
      }
      ps += __shfl_xor(ps, 1, 32);
      ps += __shfl_xor(ps, 2, 32);
      ps += __shfl_xor(ps, 4, 32);
      ps += __shfl_xor(ps, 8, 32);
      rsum[r] = rsum[r] * scale[r] + ps;
    }
#pragma unroll
    for (int ct = 0; ct < 8; ++ct)
#pragma unroll
      for (int r = 0; r < 8; ++r) oacc[ct][r] = oacc[ct][r] * scale[r];

    __syncthreads();
    if (!(wave & 1)) {                 // wave-uniform branch; EXEC stays full
#pragma unroll
      for (int mt = 0; mt < 4; ++mt)
#pragma unroll
        for (int r = 0; r < 8; ++r)
          P[(strip + r + rowoff) * 72 + mt * 16 + lrow] = (_Float16)s[mt][r];
    }
    __syncthreads();

#pragma unroll
    for (int kc = 0; kc < 2; ++kc) {
      const _Float16* pp = &P[(strip + lrow) * 72 + kc * 32];
      v16h pa = frag_ld(pp + kb0, pp + kb0 + 16);
#pragma unroll
      for (int ct = 0; ct < 8; ++ct) {
#if HAVE_TDM
        const _Float16* pv = &VL[(cbase + ct * 16 + lrow) * 64 + kc * 32];
#else
        const _Float16* pv = &gP[((size_t)b * 256 + cbase + ct * 16 + lrow) * 1024 + mb + kc * 32];
#endif
        v16h bf = frag_ld(pv + kb0, pv + kb0 + 16);
        oacc[ct] = WMMA_F16(pa, bf, oacc[ct]);
      }
    }
  }

  float inv[8];
#pragma unroll
  for (int r = 0; r < 8; ++r) inv[r] = 1.0f / rsum[r];
#pragma unroll
  for (int ct = 0; ct < 8; ++ct)
#pragma unroll
    for (int r = 0; r < 8; ++r)
      o_in[((size_t)b * 4096 + q0 + strip + r + rowoff) * 256 + cbase + ct * 16 + lrow] =
          (_Float16)(oacc[ct][r] * inv[r]);
}

// ---------------------------------------------------------------------------
// Kernel 5: output GEMM + residual.
//   out[b][k][n] = gamma * sum_c oWh[k][c] * o_in[b][n][c] + x[b][k][n]
// 512 threads = 16 waves: 8 M-strips (64 rows) x 2 N-halves (64 cols).
// Both operands read as contiguous f16 fragments from global (no LDS needed).
// ---------------------------------------------------------------------------
__global__ __launch_bounds__(512) void k_out(const _Float16* __restrict__ oWh,
                                             const _Float16* __restrict__ o_in,
                                             const float* __restrict__ x,
                                             const float* __restrict__ gamma,
                                             float* __restrict__ out) {
  const int b = blockIdx.y;
  const int n0 = blockIdx.x * 128;
  const int tid = threadIdx.x;
  const int lane = tid & 31, wave = tid >> 5;
  const int m0 = (wave >> 1) * 64;
  const int nb = (wave & 1) * 64;
  const int lrow = lane & 15;
  const int kb0 = (lane < 16) ? 0 : 8;
  const int rowoff = (lane < 16) ? 0 : 8;

  const v8f vzero = {0, 0, 0, 0, 0, 0, 0, 0};
  v8f acc[4][4];
#pragma unroll
  for (int mt = 0; mt < 4; ++mt)
#pragma unroll
    for (int nt = 0; nt < 4; ++nt) acc[mt][nt] = vzero;

  for (int kc = 0; kc < 8; ++kc) {
    v16h bfr[4];
#pragma unroll
    for (int nt = 0; nt < 4; ++nt) {
      const _Float16* p = &o_in[((size_t)b * 4096 + n0 + nb + nt * 16 + lrow) * 256 + kc * 32];
      bfr[nt] = frag_ld(p + kb0, p + kb0 + 16);
    }
#pragma unroll
    for (int mt = 0; mt < 4; ++mt) {
      const _Float16* p = &oWh[(m0 + mt * 16 + lrow) * 256 + kc * 32];
      v16h af = frag_ld(p + kb0, p + kb0 + 16);
#pragma unroll
      for (int nt = 0; nt < 4; ++nt) acc[mt][nt] = WMMA_F16(af, bfr[nt], acc[mt][nt]);
    }
  }

  const float gm = *gamma;
#pragma unroll
  for (int mt = 0; mt < 4; ++mt)
#pragma unroll
    for (int nt = 0; nt < 4; ++nt)
#pragma unroll
      for (int r = 0; r < 8; ++r) {
        const int k = m0 + mt * 16 + r + rowoff;
        const size_t idx = ((size_t)b * 512 + k) * 4096 + n0 + nb + nt * 16 + lrow;
        out[idx] = gm * acc[mt][nt][r] + x[idx];
      }
}

// ---------------------------------------------------------------------------
extern "C" void kernel_launch(void* const* d_in, const int* in_sizes, int n_in,
                              void* d_out, int out_size, void* d_ws, size_t ws_size,
                              hipStream_t stream) {
  const float* x       = (const float*)d_in[0];
  const float* theta_w = (const float*)d_in[1];
  const float* phi_w   = (const float*)d_in[2];
  const float* g_w     = (const float*)d_in[3];
  const float* o_w     = (const float*)d_in[4];
  const float* gamma   = (const float*)d_in[5];
  float* out = (float*)d_out;
  (void)in_sizes; (void)n_in; (void)out_size; (void)ws_size;

  char* ws = (char*)d_ws;
  size_t off = 0;
  auto bump = [&](size_t bytes) -> void* {
    void* p = ws + off;
    off = (off + bytes + 255) & ~(size_t)255;
    return p;
  };
  _Float16* Wc    = (_Float16*)bump((size_t)384 * 512 * 2);          // packed weights
  _Float16* oWh   = (_Float16*)bump((size_t)512 * 256 * 2);          // output weights
  _Float16* theta = (_Float16*)bump((size_t)16 * 4096 * 64 * 2);     // Q, [B][n][64]
  _Float16* pgraw = (_Float16*)bump((size_t)16 * 320 * 4096 * 2);    // pre-pool phi+g
  _Float16* phiT  = (_Float16*)bump((size_t)16 * 1024 * 64 * 2);     // K^T, [B][m][64]
  _Float16* gP    = (_Float16*)bump((size_t)16 * 256 * 1024 * 2);    // V, [B][c][m]
  _Float16* o_in  = (_Float16*)bump((size_t)16 * 4096 * 256 * 2);    // attn out, [B][n][c]

  k_convert<<<dim3(768), dim3(256), 0, stream>>>(theta_w, phi_w, g_w, o_w, Wc, oWh);
  k_proj<<<dim3(32, 16), dim3(512), 0, stream>>>(x, Wc, theta, pgraw);
  k_pool<<<dim3((16 * 320 * 1024) / 256), dim3(256), 0, stream>>>(pgraw, phiT, gP);
  k_attn<<<dim3(64, 16), dim3(256), 0, stream>>>(theta, phiT, gP, o_in);
  k_out<<<dim3(32, 16), dim3(512), 0, stream>>>(oWh, o_in, x, gamma, out);
}
